// BahdanauAttention_55954833932485
// MI455X (gfx1250) — matrix operand order repaired and checked by reading, not compiled, not run on hardware
//
#include <hip/hip_runtime.h>
#include <cmath>

// Bahdanau attention, fused/tiled for MI455X (gfx1250, wave32, WMMA + TDM).
// B=16, T=8192, D=256, U=256.

#define B_ 16
#define T_ 8192
#define D_ 256
#define U_ 256

typedef __attribute__((ext_vector_type(16))) __bf16 v16bf;
typedef __attribute__((ext_vector_type(8)))  float  v8f;

typedef __attribute__((ext_vector_type(4))) unsigned int u32x4;
typedef __attribute__((ext_vector_type(8))) int          i32x8;
typedef __attribute__((ext_vector_type(4))) int          i32x4;

#if defined(__HIP_DEVICE_COMPILE__) && __has_builtin(__builtin_amdgcn_tensor_load_to_lds)
#define USE_TDM 1
#else
#define USE_TDM 0
#endif

// ---------------------------------------------------------------------------
// Kernel 1: prep
//  blocks 0..15 : hidden[b][u] = sum_d query[b][d] * W2[d][u]
//  blocks 16..31: repack W1 (fp32 DxU) into bf16 WMMA B-fragments.
//    B-fragment (32x16, K x N) per-lane layout (wave32):
//      n = lane & 15 ; lanes 0-15 hold K=0..15, lanes 16-31 hold K=16..31
//      element j (0..15) of the v16bf  <->  k_local = (lane>>4)*16 + j
//    Stored per lane as 16 contiguous bf16 (32 bytes) for b128 loads.
//    Frag (nt,kt): base byte offset nt*8192 + kt*1024 (1KB per frag).
// ---------------------------------------------------------------------------
__global__ void __launch_bounds__(256)
bahdanau_prep_kernel(const float* __restrict__ query,
                     const float* __restrict__ W1,
                     const float* __restrict__ W2,
                     float* __restrict__ hidden,
                     v16bf* __restrict__ w1bf) {
    int blk = blockIdx.x;
    int tid = threadIdx.x;
    if (blk < 16) {
        int b = blk, u = tid;
        float s = 0.f;
        #pragma unroll 4
        for (int d = 0; d < D_; ++d)
            s += query[b * D_ + d] * W2[d * U_ + u];
        hidden[b * U_ + u] = s;
    } else {
        int nt   = blk - 16;        // N-tile 0..15 (U = 16 tiles of 16)
        int kt   = tid >> 5;        // K-tile 0..7  (D = 8 tiles of 32)
        int lane = tid & 31;
        int u     = nt * 16 + (lane & 15);
        int dbase = kt * 32 + ((lane >> 4) << 4);  // +0 or +16
        v16bf frag;
        #pragma unroll
        for (int j = 0; j < 16; ++j)
            frag[j] = (__bf16)W1[(dbase + j) * U_ + u];
        w1bf[(nt * 8 + kt) * 32 + lane] = frag;
    }
}

// ---------------------------------------------------------------------------
// TDM helper: stage one K-step worth of W1 B-fragments (16 nt x 1KB) into LDS.
//  Memory view : 2D tile, 8-byte units; tile 128 x 16, row stride 1024 units
//                (nt rows are 8KB apart in the packed w1bf array).
//  LDS view    : rows appended contiguously -> nt*1KB + lane*32B.
//  D# per CDNA5 ISA ch.8 (group0: count/lds/global/type=2; group1: dims).
//  This toolchain exposes the 6-arg builtin form:
//    (u32x4 g0, i32x8 g1, i32x4, i32x4, i32x8, i32 cpol)
// ---------------------------------------------------------------------------
#if USE_TDM
__device__ __forceinline__ void tdm_stage_w1(const void* gsrc, unsigned int lds_off) {
    unsigned long long ga = (unsigned long long)(uintptr_t)gsrc;
    u32x4 g0;
    g0[0] = 1u;                                               // count=1, no gather
    g0[1] = lds_off;                                          // lds_addr (bytes)
    g0[2] = (unsigned int)(ga & 0xffffffffull);               // global_addr[31:0]
    g0[3] = (unsigned int)((ga >> 32) & 0x01fffffful)         // global_addr[56:32]
          | (2u << 30);                                       // type=2 ("image")
    i32x8 g1;
    g1[0] = 0x00030000;                 // workgroup_mask=0, data_size=3 (8B units)
    g1[1] = (int)(128u << 16);          // tensor_dim0 = 128  (units)
    g1[2] = (int)(16u  << 16);          // tensor_dim1 = 16
    g1[3] = (int)(128u << 16);          // tile_dim0   = 128  (= 1KB per nt row)
    g1[4] = 16;                         // tile_dim1   = 16, tile_dim2 = 0
    g1[5] = 1024;                       // tensor_dim0_stride = 1024 units (8KB)
    g1[6] = 0;                          // stride0 hi / stride1 lo
    g1[7] = 0;                          // stride1 hi
    i32x4 z4 = {0, 0, 0, 0};            // groups 2/3 unused (<=2D tensor)
    i32x8 z8 = {0, 0, 0, 0, 0, 0, 0, 0};
    __builtin_amdgcn_tensor_load_to_lds(g0, g1, z4, z4, z8, 0);
}
#endif

// ---------------------------------------------------------------------------
// Kernel 2: scores[b][t] = V^T tanh(W1^T value[b,t,:] + hidden[b,:])
//  Grid (T/64, B); block = 256 threads = 8 waves.
//  Wave (mt = wave&3, nh = wave>>2): keys tile 16(T) x 128(U) via 8 K-steps of
//  v_wmma_f32_16x16x32_bf16 (8 accumulator fragments).
//  A-fragment (16x32, M x K) per-lane layout:
//      m = lane & 15
//      lanes 0-15 : K = 0..15;  lanes 16-31: K = 16..31 (same order as B)
//  -> two contiguous 8-float chunks of `value` per lane per K-step.
//  B-fragments are TDM-staged into a double-buffered 2x16KB LDS tile per
//  K-step (loaded once from L2, consumed by all 8 waves via ds_load_b128).
// ---------------------------------------------------------------------------
__global__ void __launch_bounds__(256)
bahdanau_scores_kernel(const float* __restrict__ value,
                       const float* __restrict__ hidden,
                       const float* __restrict__ Vw,
                       const v16bf* __restrict__ w1bf,
                       float* __restrict__ scores) {
    __shared__ float s_score[64];
#if USE_TDM
    __shared__ v16bf lds_w1[2][16 * 32];   // 2 x 16KB double buffer
#endif

    int b    = blockIdx.y;
    int tblk = blockIdx.x;          // 64 timesteps per block
    int tid  = threadIdx.x;
    int wave = tid >> 5;
    int lane = tid & 31;
    int mt   = wave & 3;            // M-tile within block (4 x 16 rows)
    int nh   = wave >> 2;           // 0/1: which half of U (8 N-tiles)

    if (tid < 64) s_score[tid] = 0.f;

    int trow = tblk * 64 + mt * 16 + (lane & 15);
    const float* vrow = value + ((size_t)b * T_ + trow) * D_;
    int kb_off = (lane >> 4) << 4;  // 0 or 16 (A-layout lane-half K shift)

    v8f acc[8] = {};                // 8 N-tiles of 16x16 f32

#if USE_TDM
    const char* w1base = (const char*)w1bf;
    if (wave == 0)
        tdm_stage_w1(w1base, (unsigned int)(uintptr_t)&lds_w1[0][0]);
#endif

    #pragma unroll
    for (int kt = 0; kt < 8; ++kt) {
#if USE_TDM
        if (wave == 0) {
            if (kt < 7) {
                tdm_stage_w1(w1base + (size_t)(kt + 1) * 1024,
                             (unsigned int)(uintptr_t)&lds_w1[(kt + 1) & 1][0]);
                // in-order TDM completion: <=1 outstanding => stage(kt) done
                __builtin_amdgcn_s_wait_tensorcnt(1);
            } else {
                __builtin_amdgcn_s_wait_tensorcnt(0);
            }
        }
        __syncthreads();            // publish lds_w1[kt&1] (and s_score init)
#else
        if (kt == 0) __syncthreads();
#endif
        int kb = kt * 32 + kb_off;
        const float4* p = (const float4*)(vrow + kb);
        float4 c0 = p[0];
        float4 c1 = p[1];
        const float4* q = (const float4*)(vrow + kb + 8);
        float4 c2 = q[0];
        float4 c3 = q[1];
        v16bf a;
        a[0]  = (__bf16)c0.x; a[1]  = (__bf16)c0.y; a[2]  = (__bf16)c0.z; a[3]  = (__bf16)c0.w;
        a[4]  = (__bf16)c1.x; a[5]  = (__bf16)c1.y; a[6]  = (__bf16)c1.z; a[7]  = (__bf16)c1.w;
        a[8]  = (__bf16)c2.x; a[9]  = (__bf16)c2.y; a[10] = (__bf16)c2.z; a[11] = (__bf16)c2.w;
        a[12] = (__bf16)c3.x; a[13] = (__bf16)c3.y; a[14] = (__bf16)c3.z; a[15] = (__bf16)c3.w;

        #pragma unroll
        for (int i = 0; i < 8; ++i) {
            int nt = nh * 8 + i;
#if USE_TDM
            v16bf bm = lds_w1[kt & 1][nt * 32 + lane];      // ds_load_b128 x2
#else
            v16bf bm = w1bf[(nt * 8 + kt) * 32 + lane];
#endif
            acc[i] = __builtin_amdgcn_wmma_f32_16x16x32_bf16(
                false, a, false, bm, (short)0, acc[i], false, false);
        }
#if USE_TDM
        __syncthreads();            // all waves done with lds_w1[kt&1]
#endif
    }

    // Epilogue: tanh(keys + hidden) . V  -> per-lane partial row sums.
    // C/D layout: lane holds column n = lane&15; VGPR v -> row m = v + 8*(lane>=16).
    float partial[8];
    #pragma unroll
    for (int v = 0; v < 8; ++v) partial[v] = 0.f;

    #pragma unroll
    for (int i = 0; i < 8; ++i) {
        int u = (nh * 8 + i) * 16 + (lane & 15);
        float hv = hidden[b * U_ + u];
        float vw = Vw[u];
        #pragma unroll
        for (int v = 0; v < 8; ++v)
            partial[v] += vw * tanhf(acc[i][v] + hv);
    }

    int mrow_hi = (lane & 16) ? 8 : 0;
    #pragma unroll
    for (int v = 0; v < 8; ++v)
        atomicAdd(&s_score[mt * 16 + mrow_hi + v], partial[v]);  // ds_add_f32
    __syncthreads();

    if (tid < 64)
        scores[(size_t)b * T_ + tblk * 64 + tid] = s_score[tid];
}

// ---------------------------------------------------------------------------
// Kernel 3: per-batch softmax over T, in place: scores -> weights.
// ---------------------------------------------------------------------------
__global__ void __launch_bounds__(256)
bahdanau_softmax_kernel(float* __restrict__ scores) {
    __shared__ float red[256];
    int b = blockIdx.x, tid = threadIdx.x;
    float* s = scores + (size_t)b * T_;

    float m = -3.402823466e38f;
    for (int t = tid; t < T_; t += 256) m = fmaxf(m, s[t]);
    red[tid] = m; __syncthreads();
    for (int off = 128; off > 0; off >>= 1) {
        if (tid < off) red[tid] = fmaxf(red[tid], red[tid + off]);
        __syncthreads();
    }
    float bmax = red[0];
    __syncthreads();

    float loc[T_ / 256];
    float sum = 0.f;
    #pragma unroll
    for (int i = 0; i < T_ / 256; ++i) {
        float e = expf(s[tid + i * 256] - bmax);
        loc[i] = e;
        sum += e;
    }
    red[tid] = sum; __syncthreads();
    for (int off = 128; off > 0; off >>= 1) {
        if (tid < off) red[tid] += red[tid + off];
        __syncthreads();
    }
    float inv = 1.f / red[0];
    #pragma unroll
    for (int i = 0; i < T_ / 256; ++i)
        s[tid + i * 256] = loc[i] * inv;
}

// ---------------------------------------------------------------------------
// Kernel 4: context[b][d] = sum_t weights[b][t] * value[b][t][d]
//  Grid (D/64, B); block 256: 64 d-columns x 4 T-slices. value re-read from L2
//  (134 MB < 192 MB L2). Coalesced in d.
// ---------------------------------------------------------------------------
__global__ void __launch_bounds__(256)
bahdanau_context_kernel(const float* __restrict__ value,
                        const float* __restrict__ weights,
                        float* __restrict__ out) {
    __shared__ float red[256];
    int b     = blockIdx.y;
    int dbase = blockIdx.x * 64;
    int tid   = threadIdx.x;
    int d     = dbase + (tid & 63);
    int slice = tid >> 6;

    const float* w  = weights + (size_t)b * T_;
    const float* vb = value   + (size_t)b * T_ * D_;

    float acc = 0.f;
    #pragma unroll 4
    for (int t = slice; t < T_; t += 4)
        acc += w[t] * vb[(size_t)t * D_ + d];

    red[tid] = acc; __syncthreads();
    if (tid < 64)
        out[b * D_ + dbase + tid] =
            red[tid] + red[tid + 64] + red[tid + 128] + red[tid + 192];
}

// ---------------------------------------------------------------------------
extern "C" void kernel_launch(void* const* d_in, const int* in_sizes, int n_in,
                              void* d_out, int out_size, void* d_ws, size_t ws_size,
                              hipStream_t stream) {
    (void)in_sizes; (void)n_in; (void)out_size; (void)ws_size;
    const float* query = (const float*)d_in[0];   // [B, D]
    const float* value = (const float*)d_in[1];   // [B, T, D]
    const float* W1    = (const float*)d_in[2];   // [D, U]
    const float* W2    = (const float*)d_in[3];   // [D, U]
    const float* Vw    = (const float*)d_in[4];   // [U]
    float* out = (float*)d_out;                   // [B, D]

    // Workspace layout (all recomputed every launch; no cross-call state):
    //   [0,      16384)  hidden  (B*U f32)
    //   [16384, 147456)  W1 bf16 B-fragments (16 nt * 8 kt * 32 lanes * 32B)
    //   [147456, ...  )  scores/weights (B*T f32 = 512 KB)
    char*  ws      = (char*)d_ws;
    float* hidden  = (float*)ws;
    v16bf* w1bf    = (v16bf*)(ws + 16384);
    float* scores  = (float*)(ws + 16384 + 16 * 8 * 32 * (int)sizeof(v16bf));

    bahdanau_prep_kernel<<<32, 256, 0, stream>>>(query, W1, W2, hidden, w1bf);
    bahdanau_scores_kernel<<<dim3(T_ / 64, B_), 256, 0, stream>>>(
        value, hidden, Vw, w1bf, scores);
    bahdanau_softmax_kernel<<<B_, 256, 0, stream>>>(scores);
    bahdanau_context_kernel<<<dim3(D_ / 64, B_), 256, 0, stream>>>(
        value, scores, out);
}
